// TransformerBlockLite_64304250356103
// MI455X (gfx1250) — matrix operand order repaired and checked by reading, not compiled, not run on hardware
//
#include <hip/hip_runtime.h>
#include <hip/hip_bf16.h>
#include <math.h>

// ---------------- problem constants ----------------
#define Bn   16
#define Cc   256
#define Hh   64
#define Ww   64
#define Nn   4096          // H*W
#define Pp   65536         // B*N pixels
#define HEADS 4
#define KD   32
#define VD   32
#define HID  512
#define EPSL 1e-5f

typedef __attribute__((ext_vector_type(16))) _Float16 v16h;
typedef __attribute__((ext_vector_type(8)))  _Float16 v8h;
typedef __attribute__((ext_vector_type(8)))  float    v8f;

// ---------------- wave32 reductions ----------------
__device__ __forceinline__ float wred_add(float v) {
#pragma unroll
  for (int off = 16; off > 0; off >>= 1) v += __shfl_xor(v, off, 32);
  return v;
}
__device__ __forceinline__ float wred_max(float v) {
#pragma unroll
  for (int off = 16; off > 0; off >>= 1) v = fmaxf(v, __shfl_xor(v, off, 32));
  return v;
}

// ------------- CDNA5 async copy to LDS -------------
__device__ __forceinline__ void async_load_b128(unsigned lds_off, const void* gaddr) {
  // VDST = LDS byte address, VADDR = 64-bit global address (tracked by ASYNCcnt)
  asm volatile("global_load_async_to_lds_b128 %0, %1, off"
               :: "v"(lds_off), "v"(gaddr) : "memory");
}
__device__ __forceinline__ void wait_async0() {
  asm volatile("s_wait_asynccnt 0x0" ::: "memory");
}

// ---------------- WMMA macro-tile core -------------
// Block = 256 threads (8 waves) computes M=128 x N=64.
// Wave w: mw = w&3 (rows mw*32..+31), nw = w>>2 (cols nw*32..+31),
// each wave holds a 2x2 grid of 16x16 fp32 accumulators.
// A (activations) [P,K] f16 staged in LDS via async loads, double buffered.
// B (weights)     [O,K] f16 read directly (L2 resident).
template <int K>
__device__ __forceinline__ void gemm_block(const _Float16* __restrict__ A,
                                           const _Float16* __restrict__ W,
                                           long mBlock, long nBlock,
                                           v8f acc[2][2]) {
  __shared__ _Float16 sA[2][128][40];   // 80B row pitch (16B-aligned), 20KB
  const int tid  = threadIdx.x;
  const int lane = tid & 31;
  const int wave = tid >> 5;
  const int mw = wave & 3, nw = wave >> 2;
  const int nlane = lane & 15;
  const int kA = (lane < 16) ? 0 : 16;    // A half-row select (halfs)
  const int kB = (lane < 16) ? 0 : 16;    // B K-half select (halfs)

  auto stage = [&](int buf, int k0) {
    // 128 rows x 64B = 512 x 16B chunks; 2 per thread
#pragma unroll
    for (int rep = 0; rep < 2; ++rep) {
      int c = tid + rep * 256;
      int row = c >> 2;
      int off = (c & 3) * 8;              // halfs within row
      unsigned lds = (unsigned)(unsigned long long)(&sA[buf][row][off]);
      const void* g = (const void*)(A + (mBlock + row) * (long)K + k0 + off);
      async_load_b128(lds, g);
    }
  };

  stage(0, 0);
  int buf = 0;
#pragma unroll
  for (int k0 = 0; k0 < K; k0 += 32) {
    wait_async0();          // my async stores to LDS landed
    __syncthreads();        // everyone's landed
    if (k0 + 32 < K) stage(buf ^ 1, k0 + 32);

    // B fragments (2 N-tiles) straight from global
    v16h bfr[2];
#pragma unroll
    for (int nt = 0; nt < 2; ++nt) {
      long col = nBlock + nw * 32 + nt * 16 + nlane;
      bfr[nt] = *(const v16h*)(W + col * (long)K + k0 + kB);
    }
    // A fragments (2 M-tiles) from LDS, 2x2 WMMA
#pragma unroll
    for (int mt = 0; mt < 2; ++mt) {
      int row = mw * 32 + mt * 16 + nlane;
      v8h alo = *(const v8h*)&sA[buf][row][kA];
      v8h ahi = *(const v8h*)&sA[buf][row][kA + 8];
      v16h a;
#pragma unroll
      for (int i = 0; i < 8; ++i) { a[i] = alo[i]; a[i + 8] = ahi[i]; }
#pragma unroll
      for (int nt = 0; nt < 2; ++nt)
        acc[mt][nt] = __builtin_amdgcn_wmma_f32_16x16x32_f16(
            false, a, false, bfr[nt], (short)0, acc[mt][nt], false, false);
    }
    __syncthreads();        // reads done before buffer is overwritten
    buf ^= 1;
  }
}

// ---------------- fp32 -> f16 convert --------------
__global__ void k_f2h(const float* __restrict__ s, _Float16* __restrict__ d, int n) {
  int i = blockIdx.x * 256 + threadIdx.x;
  if (i < n) d[i] = (_Float16)s[i];
}

// ---------------- LN over channels of NCHW ---------
__global__ void k_ln1(const float* __restrict__ x, const float* __restrict__ w,
                      const float* __restrict__ bb, _Float16* __restrict__ xn) {
  long b  = blockIdx.x >> 4;
  int hw  = ((blockIdx.x & 15) << 8) + threadIdx.x;
  const float* xp = x + b * (long)Cc * Nn + hw;
  float s = 0.f, ss = 0.f;
  for (int c = 0; c < Cc; ++c) { float v = xp[(long)c * Nn]; s += v; ss += v * v; }
  float mean = s * (1.f / Cc);
  float var  = ss * (1.f / Cc) - mean * mean;
  float rstd = rsqrtf(var + EPSL);
  long p = b * Nn + hw;
  for (int c = 0; c < Cc; ++c) {
    float v = xp[(long)c * Nn];
    xn[p * Cc + c] = (_Float16)((v - mean) * rstd * w[c] + bb[c]);
  }
}

// LN over contiguous rows [P,256] (wave per row)
__global__ void k_ln2(const float* __restrict__ x1, const float* __restrict__ w,
                      const float* __restrict__ bb, _Float16* __restrict__ xn) {
  int lane = threadIdx.x & 31;
  long p = (long)blockIdx.x * 8 + (threadIdx.x >> 5);
  const float* xp = x1 + p * Cc;
  float vals[8]; float s = 0.f;
#pragma unroll
  for (int j = 0; j < 8; ++j) { vals[j] = xp[lane + 32 * j]; s += vals[j]; }
  float mean = wred_add(s) * (1.f / Cc);
  float sq = 0.f;
#pragma unroll
  for (int j = 0; j < 8; ++j) { float d = vals[j] - mean; sq += d * d; }
  float rstd = rsqrtf(wred_add(sq) * (1.f / Cc) + EPSL);
#pragma unroll
  for (int j = 0; j < 8; ++j) {
    int c = lane + 32 * j;
    xn[p * Cc + c] = (_Float16)((vals[j] - mean) * rstd * w[c] + bb[c]);
  }
}

// ---------------- GEMM kernels ---------------------
__global__ void __launch_bounds__(256)
k_gemm_qkv(const _Float16* __restrict__ xn, const _Float16* __restrict__ wqkv,
           const float* __restrict__ bq, const float* __restrict__ bk,
           const float* __restrict__ bv, float* __restrict__ qkv) {
  v8f acc[2][2] = {};
  long m0 = (long)blockIdx.x * 128, n0 = (long)blockIdx.y * 64;
  gemm_block<256>(xn, wqkv, m0, n0, acc);
  const int lane = threadIdx.x & 31, wave = threadIdx.x >> 5;
  const int mw = wave & 3, nw = wave >> 2;
  const int mBh = (lane < 16) ? 0 : 8;
#pragma unroll
  for (int nt = 0; nt < 2; ++nt) {
    int n = (int)n0 + nw * 32 + nt * 16 + (lane & 15);
    float bias = (n < 128) ? bq[n] : (n < 256) ? bk[n - 128] : bv[n - 256];
#pragma unroll
    for (int mt = 0; mt < 2; ++mt)
#pragma unroll
      for (int r = 0; r < 8; ++r) {
        long p = m0 + mw * 32 + mt * 16 + mBh + r;
        qkv[p * 384 + n] = acc[mt][nt][r] + bias;
      }
  }
}

__global__ void __launch_bounds__(256)
k_gemm_wo(const _Float16* __restrict__ oh, const _Float16* __restrict__ woh,
          const float* __restrict__ bo, const float* __restrict__ x,
          float* __restrict__ x1) {
  v8f acc[2][2] = {};
  long m0 = (long)blockIdx.x * 128, n0 = (long)blockIdx.y * 64;
  gemm_block<128>(oh, woh, m0, n0, acc);
  const int lane = threadIdx.x & 31, wave = threadIdx.x >> 5;
  const int mw = wave & 3, nw = wave >> 2;
  const int mBh = (lane < 16) ? 0 : 8;
#pragma unroll
  for (int nt = 0; nt < 2; ++nt) {
    int n = (int)n0 + nw * 32 + nt * 16 + (lane & 15);
    float bias = bo[n];
#pragma unroll
    for (int mt = 0; mt < 2; ++mt)
#pragma unroll
      for (int r = 0; r < 8; ++r) {
        long p = m0 + mw * 32 + mt * 16 + mBh + r;
        long b = p >> 12, hw = p & 4095;
        x1[p * Cc + n] = acc[mt][nt][r] + bias + x[(b * Cc + n) * (long)Nn + hw];
      }
  }
}

__global__ void __launch_bounds__(256)
k_gemm_pw1(const _Float16* __restrict__ xn2, const _Float16* __restrict__ w1h,
           _Float16* __restrict__ u) {
  v8f acc[2][2] = {};
  long m0 = (long)blockIdx.x * 128, n0 = (long)blockIdx.y * 64;
  gemm_block<256>(xn2, w1h, m0, n0, acc);
  const int lane = threadIdx.x & 31, wave = threadIdx.x >> 5;
  const int mw = wave & 3, nw = wave >> 2;
  const int mBh = (lane < 16) ? 0 : 8;
#pragma unroll
  for (int nt = 0; nt < 2; ++nt) {
    int n = (int)n0 + nw * 32 + nt * 16 + (lane & 15);
#pragma unroll
    for (int mt = 0; mt < 2; ++mt)
#pragma unroll
      for (int r = 0; r < 8; ++r) {
        long p = m0 + mw * 32 + mt * 16 + mBh + r;
        u[p * 1024 + n] = (_Float16)acc[mt][nt][r];
      }
  }
}

__global__ void __launch_bounds__(256)
k_gemm_pw2(const _Float16* __restrict__ mh, const _Float16* __restrict__ w2h,
           const float* __restrict__ x1, float* __restrict__ out) {
  v8f acc[2][2] = {};
  long m0 = (long)blockIdx.x * 128, n0 = (long)blockIdx.y * 64;
  gemm_block<512>(mh, w2h, m0, n0, acc);
  const int lane = threadIdx.x & 31, wave = threadIdx.x >> 5;
  const int mw = wave & 3, nw = wave >> 2;
  const int mBh = (lane < 16) ? 0 : 8;
#pragma unroll
  for (int nt = 0; nt < 2; ++nt) {
    int n = (int)n0 + nw * 32 + nt * 16 + (lane & 15);
#pragma unroll
    for (int mt = 0; mt < 2; ++mt)
#pragma unroll
      for (int r = 0; r < 8; ++r) {
        long p = m0 + mw * 32 + mt * 16 + mBh + r;
        long b = p >> 12, hw = p & 4095;
        out[(b * Cc + n) * (long)Nn + hw] = acc[mt][nt][r] + x1[p * Cc + n];
      }
  }
}

// ---------------- attention small ops --------------
__global__ void k_attn_norm(const float* __restrict__ qkv,
                            const float* __restrict__ nqw, const float* __restrict__ nqb,
                            const float* __restrict__ nkw, const float* __restrict__ nkb,
                            const float* __restrict__ nvw, const float* __restrict__ nvb,
                            float* __restrict__ qs, float* __restrict__ klns,
                            float* __restrict__ vln) {
  int lane = threadIdx.x & 31;
  long vid = (long)blockIdx.x * 8 + (threadIdx.x >> 5);   // bh*4096 + n
  int bh = (int)(vid >> 12), n = (int)(vid & 4095);
  int b = bh >> 2, h = bh & 3;
  const float* base = qkv + ((long)b * Nn + n) * 384;
  float qv = base[h * 32 + lane];
  float kv = base[128 + h * 32 + lane];
  float vv = base[256 + h * 32 + lane];
  const float scale = 0.42044820762685725f;   // 32^-0.25
  auto ln = [&](float v, const float* w, const float* bia) {
    float m = wred_add(v) * (1.f / 32.f);
    float d = v - m;
    float var = wred_add(d * d) * (1.f / 32.f);
    return d * rsqrtf(var + EPSL) * w[lane] + bia[lane];
  };
  qv = ln(qv, nqw, nqb) * scale;
  kv = ln(kv, nkw, nkb) * scale;
  vv = ln(vv, nvw, nvb);
  float qm = wred_max(qv);
  float qe = __expf(qv - qm);
  float qz = wred_add(qe);
  long o = vid * 32 + lane;
  qs[o]   = qe / qz;
  klns[o] = kv;
  vln[o]  = vv;
}

__global__ void k_ksoft_stats(const float* __restrict__ klns,
                              float* __restrict__ kmax, float* __restrict__ ksum) {
  __shared__ float sm[256];
  int bh = blockIdx.x >> 5, d = blockIdx.x & 31;
  const float* kp = klns + (long)bh * Nn * 32 + d;
  float mx = -1e30f;
  for (int n = threadIdx.x; n < Nn; n += 256) mx = fmaxf(mx, kp[(long)n * 32]);
  sm[threadIdx.x] = mx; __syncthreads();
  for (int s = 128; s > 0; s >>= 1) {
    if (threadIdx.x < s) sm[threadIdx.x] = fmaxf(sm[threadIdx.x], sm[threadIdx.x + s]);
    __syncthreads();
  }
  mx = sm[0]; __syncthreads();
  float su = 0.f;
  for (int n = threadIdx.x; n < Nn; n += 256) su += __expf(kp[(long)n * 32] - mx);
  sm[threadIdx.x] = su; __syncthreads();
  for (int s = 128; s > 0; s >>= 1) {
    if (threadIdx.x < s) sm[threadIdx.x] += sm[threadIdx.x + s];
    __syncthreads();
  }
  if (threadIdx.x == 0) { kmax[blockIdx.x] = mx; ksum[blockIdx.x] = sm[0]; }
}

__global__ void k_ctx(const float* __restrict__ klns, const float* __restrict__ vln,
                      const float* __restrict__ kmax, const float* __restrict__ ksum,
                      float* __restrict__ ctx) {
  int bh = blockIdx.x;
  int d = threadIdx.x >> 5, e = threadIdx.x & 31;
  const float* kp = klns + (long)bh * Nn * 32 + d;
  const float* vp = vln  + (long)bh * Nn * 32 + e;
  float mx = kmax[bh * 32 + d];
  float inv = 1.f / ksum[bh * 32 + d];
  float acc = 0.f;
  for (int n = 0; n < Nn; ++n)
    acc += __expf(kp[(long)n * 32] - mx) * vp[(long)n * 32];
  ctx[((long)bh * 32 + d) * 32 + e] = acc * inv;
}

__global__ void k_attn_out(const float* __restrict__ qs, const float* __restrict__ ctx,
                           _Float16* __restrict__ oh) {
  int lane = threadIdx.x & 31;
  long vid = (long)blockIdx.x * 8 + (threadIdx.x >> 5);
  int bh = (int)(vid >> 12), n = (int)(vid & 4095);
  int b = bh >> 2, h = bh & 3;
  const float* qp = qs + vid * 32;
  const float* cp = ctx + (long)bh * 1024;
  float acc = 0.f;
#pragma unroll
  for (int d = 0; d < 32; ++d) acc += qp[d] * cp[d * 32 + lane];
  oh[((long)b * Nn + n) * 128 + h * 32 + lane] = (_Float16)acc;
}

// depthwise 3x3 + gated exact-gelu; thread per (p, ch<512)
__global__ void k_dwconv(const _Float16* __restrict__ u, const float* __restrict__ dww,
                         _Float16* __restrict__ mo) {
  long idx = (long)blockIdx.x * 256 + threadIdx.x;    // P*512
  int ch = (int)(idx & 511);
  long p  = idx >> 9;
  int b = (int)(p >> 12); int hw = (int)(p & 4095);
  int y = hw >> 6, xx = hw & 63;
  const float* wA = dww + (long)ch * 9;
  const float* wG = dww + (long)(512 + ch) * 9;
  float wa = 0.f, wg = 0.f;
#pragma unroll
  for (int dy = -1; dy <= 1; ++dy)
#pragma unroll
    for (int dx = -1; dx <= 1; ++dx) {
      int yy = y + dy, xq = xx + dx;
      if (yy < 0 || yy > 63 || xq < 0 || xq > 63) continue;
      long pp = ((long)b * Nn + yy * 64 + xq) * 1024;
      int wi = (dy + 1) * 3 + (dx + 1);
      wa += wA[wi] * (float)u[pp + ch];
      wg += wG[wi] * (float)u[pp + 512 + ch];
    }
  float gel = 0.5f * wg * (1.f + erff(wg * 0.70710678118654752f));
  mo[idx] = (_Float16)(wa * gel);
}

// ---------------- host launch ----------------------
extern "C" void kernel_launch(void* const* d_in, const int* in_sizes, int n_in,
                              void* d_out, int out_size, void* d_ws, size_t ws_size,
                              hipStream_t stream) {
  const float* x     = (const float*)d_in[0];
  const float* ln1w  = (const float*)d_in[1];
  const float* ln1b  = (const float*)d_in[2];
  const float* ln2w  = (const float*)d_in[3];
  const float* ln2b  = (const float*)d_in[4];
  const float* wq    = (const float*)d_in[5];
  const float* bq    = (const float*)d_in[6];
  const float* wk    = (const float*)d_in[7];
  const float* bk    = (const float*)d_in[8];
  const float* wv    = (const float*)d_in[9];
  const float* bv    = (const float*)d_in[10];
  const float* wo    = (const float*)d_in[11];
  const float* bo    = (const float*)d_in[12];
  const float* nqw   = (const float*)d_in[13];
  const float* nqb   = (const float*)d_in[14];
  const float* nkw   = (const float*)d_in[15];
  const float* nkb   = (const float*)d_in[16];
  const float* nvw   = (const float*)d_in[17];
  const float* nvb   = (const float*)d_in[18];
  const float* pw1   = (const float*)d_in[19];
  const float* dww   = (const float*)d_in[20];
  const float* pw2   = (const float*)d_in[21];
  float* out = (float*)d_out;

  char* cur = (char*)d_ws;
  auto carve = [&](size_t bytes) -> void* {
    void* p = (void*)cur;
    cur += (bytes + 255) & ~(size_t)255;
    return p;
  };
  _Float16* wqkv_h = (_Float16*)carve((size_t)384 * 256 * 2);
  _Float16* wo_h   = (_Float16*)carve((size_t)256 * 128 * 2);
  _Float16* w1_h   = (_Float16*)carve((size_t)1024 * 256 * 2);
  _Float16* w2_h   = (_Float16*)carve((size_t)256 * 512 * 2);
  _Float16* xn_h   = (_Float16*)carve((size_t)Pp * 256 * 2);
  float*    qkv_f  = (float*)   carve((size_t)Pp * 384 * 4);
  float*    qs_f   = (float*)   carve((size_t)64 * Nn * 32 * 4);
  float*    kln_f  = (float*)   carve((size_t)64 * Nn * 32 * 4);
  float*    vln_f  = (float*)   carve((size_t)64 * Nn * 32 * 4);
  float*    kmax_f = (float*)   carve((size_t)2048 * 4);
  float*    ksum_f = (float*)   carve((size_t)2048 * 4);
  float*    ctx_f  = (float*)   carve((size_t)64 * 1024 * 4);
  _Float16* o_h    = (_Float16*)carve((size_t)Pp * 128 * 2);
  float*    x1_f   = (float*)   carve((size_t)Pp * 256 * 4);
  _Float16* xn2_h  = (_Float16*)carve((size_t)Pp * 256 * 2);
  _Float16* u_h    = (_Float16*)carve((size_t)Pp * 1024 * 2);
  _Float16* m_h    = (_Float16*)carve((size_t)Pp * 512 * 2);

  auto cvt = [&](const float* s, _Float16* d, int n) {
    k_f2h<<<(n + 255) / 256, 256, 0, stream>>>(s, d, n);
  };
  cvt(wq,  wqkv_h,               128 * 256);
  cvt(wk,  wqkv_h + 128 * 256,   128 * 256);
  cvt(wv,  wqkv_h + 256 * 256,   128 * 256);
  cvt(wo,  wo_h,                 256 * 128);
  cvt(pw1, w1_h,                1024 * 256);
  cvt(pw2, w2_h,                 256 * 512);

  // 1) LN1 -> xn f16 [P,256]
  k_ln1<<<Bn * 16, 256, 0, stream>>>(x, ln1w, ln1b, xn_h);

  // 2) QKV GEMM (WMMA + async-LDS staging)
  k_gemm_qkv<<<dim3(Pp / 128, 384 / 64), 256, 0, stream>>>(xn_h, wqkv_h, bq, bk, bv, qkv_f);

  // 3..6) linear attention small ops
  k_attn_norm<<<(64 * Nn) / 8, 256, 0, stream>>>(qkv_f, nqw, nqb, nkw, nkb, nvw, nvb,
                                                 qs_f, kln_f, vln_f);
  k_ksoft_stats<<<2048, 256, 0, stream>>>(kln_f, kmax_f, ksum_f);
  k_ctx<<<64, 1024, 0, stream>>>(kln_f, vln_f, kmax_f, ksum_f, ctx_f);
  k_attn_out<<<(64 * Nn) / 8, 256, 0, stream>>>(qs_f, ctx_f, o_h);

  // 7) output projection + bias + residual
  k_gemm_wo<<<dim3(Pp / 128, 256 / 64), 256, 0, stream>>>(o_h, wo_h, bo, x, x1_f);

  // 8) LN2
  k_ln2<<<Pp / 8, 256, 0, stream>>>(x1_f, ln2w, ln2b, xn2_h);

  // 9) pw1 GEMM
  k_gemm_pw1<<<dim3(Pp / 128, 1024 / 64), 256, 0, stream>>>(xn2_h, w1_h, u_h);

  // 10) depthwise 3x3 + gelu gate
  k_dwconv<<<(Pp * 512) / 256, 256, 0, stream>>>(u_h, dww, m_h);

  // 11) pw2 GEMM + residual -> NCHW out
  k_gemm_pw2<<<dim3(Pp / 128, 256 / 64), 256, 0, stream>>>(m_h, w2_h, x1_f, out);

  (void)in_sizes; (void)n_in; (void)out_size; (void)ws_size;
}
